// StructureAwareAttention_18803366822083
// MI455X (gfx1250) — matrix operand order repaired and checked by reading, not compiled, not run on hardware
//
#include <hip/hip_runtime.h>

#define D_MODEL 1024
#define N_HEADS 16
#define HEAD_DIM 64
#define SEQ_L 2048

typedef unsigned short u16;
typedef unsigned int u32;

typedef __bf16 bf16_t;
typedef bf16_t v16bf __attribute__((ext_vector_type(16)));
typedef float v8f __attribute__((ext_vector_type(8)));
typedef u16 v8u16 __attribute__((ext_vector_type(8)));

union BFrag {
    v8u16 h[2];
    v16bf v;
};

__device__ inline u16 f32_to_bf16(float f) {
    u32 u = __float_as_uint(f);
    u32 r = u + 0x7FFFu + ((u >> 16) & 1u);
    return (u16)(r >> 16);
}

// --- CDNA5 async global->LDS copy (ASYNCcnt-tracked), GV addressing mode ---
__device__ inline void async_load_b128(u32 ldsoff, const float* gaddr) {
    asm volatile("global_load_async_to_lds_b128 %0, %1, off"
                 :: "v"(ldsoff), "v"(gaddr)
                 : "memory");
}
__device__ inline void wait_async_le4() {
    asm volatile("s_wait_asynccnt 4" ::: "memory");
}
__device__ inline void wait_async_le0() {
    asm volatile("s_wait_asynccnt 0" ::: "memory");
}

// ---------------------------------------------------------------------------
// Stage 1: f32 -> bf16 conversion
// ---------------------------------------------------------------------------
__global__ void convert_f32_to_bf16(const float* __restrict__ src,
                                    u16* __restrict__ dst, int n) {
    int i = blockIdx.x * blockDim.x + threadIdx.x;
    int stride = gridDim.x * blockDim.x;
    for (; i < n; i += stride) dst[i] = f32_to_bf16(src[i]);
}

// ---------------------------------------------------------------------------
// Shared GEMM core: one wave computes 16 rows x 64 cols, K = D_MODEL.
// Software-pipelined: fragments for k-step n+1 are loaded while WMMAs
// consume k-step n (double-buffered in registers; unroll-2 makes the
// buffer index compile-time).
// ---------------------------------------------------------------------------
__device__ inline void gemm_row16_col64(const u16* __restrict__ arow,
                                        const u16* __restrict__ wb,
                                        int col0, int l16, int hf,
                                        v8f acc[4]) {
    BFrag a[2];
    BFrag b[2][4];

    a[0].h[0] = *(const v8u16*)(arow + hf * 16);
    a[0].h[1] = *(const v8u16*)(arow + hf * 16 + 8);
#pragma unroll
    for (int t = 0; t < 4; ++t) {
        const u16* br = wb + (size_t)(col0 + t * 16 + l16) * D_MODEL + hf * 16;
        b[0][t].h[0] = *(const v8u16*)(br);
        b[0][t].h[1] = *(const v8u16*)(br + 8);
    }

#pragma unroll 2
    for (int k0 = 0; k0 < D_MODEL; k0 += 32) {
        const int cur = (k0 >> 5) & 1;
        const int nxt = cur ^ 1;
        const int k1 = k0 + 32;
        if (k1 < D_MODEL) {
            a[nxt].h[0] = *(const v8u16*)(arow + k1 + hf * 16);
            a[nxt].h[1] = *(const v8u16*)(arow + k1 + hf * 16 + 8);
#pragma unroll
            for (int t = 0; t < 4; ++t) {
                const u16* br =
                    wb + (size_t)(col0 + t * 16 + l16) * D_MODEL + k1 + hf * 16;
                b[nxt][t].h[0] = *(const v8u16*)(br);
                b[nxt][t].h[1] = *(const v8u16*)(br + 8);
            }
        }
#pragma unroll
        for (int t = 0; t < 4; ++t)
            acc[t] = __builtin_amdgcn_wmma_f32_16x16x32_bf16(
                false, a[cur].v, false, b[cur][t].v, (short)0, acc[t], false,
                false);
    }
}

// ---------------------------------------------------------------------------
// Stage 2: QKV projection. C[2048,3072] = X @ Wqkv^T + b, routed into
// q[h][L][64], k[h][L][64], vT[h][64][L] (bf16).
// ---------------------------------------------------------------------------
__global__ __launch_bounds__(128) void qkv_gemm_kernel(
    const u16* __restrict__ xb, const u16* __restrict__ wb,
    const float* __restrict__ bias, u16* __restrict__ qh,
    u16* __restrict__ kh, u16* __restrict__ vT) {
    const int lane = threadIdx.x & 31;
    const int wave = threadIdx.x >> 5;
    const int hf = lane >> 4;
    const int l16 = lane & 15;
    const int row0 = blockIdx.y * 64 + wave * 16;
    const int col0 = blockIdx.x * 64;

    v8f acc[4] = {};
    gemm_row16_col64(xb + (size_t)(row0 + l16) * D_MODEL, wb, col0, l16, hf,
                     acc);

#pragma unroll
    for (int t = 0; t < 4; ++t) {
        const int col = col0 + t * 16 + l16;
        const int which = col >> 10;  // 0=q 1=k 2=v
        const int head = (col >> 6) & 15;
        const int d = col & 63;
        const float bv = bias[col];
#pragma unroll
        for (int r = 0; r < 8; ++r) {
            const int row = row0 + r + hf * 8;
            const u16 val = f32_to_bf16(acc[t][r] + bv);
            if (which == 0)
                qh[((size_t)head * SEQ_L + row) * HEAD_DIM + d] = val;
            else if (which == 1)
                kh[((size_t)head * SEQ_L + row) * HEAD_DIM + d] = val;
            else
                vT[((size_t)head * HEAD_DIM + d) * SEQ_L + row] = val;
        }
    }
}

// ---------------------------------------------------------------------------
// Stage 3: flash attention per head with gated pair bias.
// One wave = 16-row query tile; j streamed in tiles of 32.
//  - K fragments double-buffered in registers (prefetch j+32)
//  - pair_bias tiles streamed via async global->LDS double buffer
//  - V fragments issued early each iteration, consumed after softmax
// ---------------------------------------------------------------------------
struct KFrag {
    BFrag kb[2][2];  // [sub][k-chunk]
};

__device__ inline void load_kfrag(KFrag& f, const u16* khH, int j0, int hf,
                                  int l16) {
#pragma unroll
    for (int sub = 0; sub < 2; ++sub) {
        const u16* krow = khH + (size_t)(j0 + sub * 16 + l16) * HEAD_DIM +
                          hf * 16;
        f.kb[sub][0].h[0] = *(const v8u16*)(krow);
        f.kb[sub][0].h[1] = *(const v8u16*)(krow + 8);
        f.kb[sub][1].h[0] = *(const v8u16*)(krow + 32);
        f.kb[sub][1].h[1] = *(const v8u16*)(krow + 40);
    }
}

__global__ __launch_bounds__(128) void attn_kernel(
    const u16* __restrict__ qh, const u16* __restrict__ kh,
    const u16* __restrict__ vT, const float* __restrict__ pair_bias,
    const float* __restrict__ gate, u16* __restrict__ attn_out) {
    __shared__ __align__(16) u16 plds[4][16 * 32];
    __shared__ __align__(16) float blds[4][2][16 * 36];  // padded rows

    const int lane = threadIdx.x & 31;
    const int wave = threadIdx.x >> 5;
    const int hf = lane >> 4;
    const int l16 = lane & 15;
    const int head = blockIdx.y;
    const int i0 = (blockIdx.x * 4 + wave) * 16;
    const float scale = 0.125f;  // 1/sqrt(64)
    const float g = gate[head];

    const u16* khH = kh + (size_t)head * SEQ_L * HEAD_DIM;
    const u16* vTH = vT + (size_t)head * HEAD_DIM * SEQ_L;
    const float* biasH = pair_bias + (size_t)head * SEQ_L * SEQ_L;

    // Q fragments, K-chunks d=[0,32) and [32,64)
    const u16* qrow = qh + ((size_t)head * SEQ_L + i0 + l16) * HEAD_DIM;
    BFrag qa0, qa1;
    qa0.h[0] = *(const v8u16*)(qrow + hf * 16);
    qa0.h[1] = *(const v8u16*)(qrow + hf * 16 + 8);
    qa1.h[0] = *(const v8u16*)(qrow + 32 + hf * 16);
    qa1.h[1] = *(const v8u16*)(qrow + 40 + hf * 16);

    v8f acc[4] = {};
    float mrow[8], lrow[8];
#pragma unroll
    for (int r = 0; r < 8; ++r) { mrow[r] = -1e30f; lrow[r] = 0.0f; }

    u16* pw = plds[wave];
    const int arow_ = lane >> 3;       // 0..3
    const int acs = (lane & 7) * 4;    // 16B col segment

    // ---- prologue: K frags for tile 0; async bias tile 0 into buf 0 ----
    KFrag kf[2];
    load_kfrag(kf[0], khH, 0, hf, l16);
#pragma unroll
    for (int ss = 0; ss < 4; ++ss) {
        const int r = ss * 4 + arow_;
        async_load_b128((u32)(size_t)&blds[wave][0][r * 36 + acs],
                        biasH + (size_t)(i0 + r) * SEQ_L + acs);
    }

#pragma unroll 2
    for (int j0 = 0; j0 < SEQ_L; j0 += 32) {
        const int cur = (j0 >> 5) & 1;
        const int nxt = cur ^ 1;
        const bool has_next = (j0 + 32 < SEQ_L);

        // ---- pipeline: prefetch next K frags + next bias tile ----
        if (has_next) {
            load_kfrag(kf[nxt], khH, j0 + 32, hf, l16);
#pragma unroll
            for (int ss = 0; ss < 4; ++ss) {
                const int r = ss * 4 + arow_;
                async_load_b128(
                    (u32)(size_t)&blds[wave][nxt][r * 36 + acs],
                    biasH + (size_t)(i0 + r) * SEQ_L + (j0 + 32) + acs);
            }
            wait_async_le4();  // current buffer complete; next 4 in flight
        } else {
            wait_async_le0();
        }
        if (j0 + 64 < SEQ_L)
            __builtin_prefetch(
                biasH + (size_t)(i0 + hf * 8) * SEQ_L + (j0 + 64 + l16), 0, 0);

        // ---- issue V loads early (consumed after softmax) ----
        BFrag vb[4];
#pragma unroll
        for (int t = 0; t < 4; ++t) {
            const u16* vrow =
                vTH + (size_t)(t * 16 + l16) * SEQ_L + j0 + hf * 16;
            vb[t].h[0] = *(const v8u16*)(vrow);
            vb[t].h[1] = *(const v8u16*)(vrow + 8);
        }

        // ---- S = Q K^T ----
        v8f s[2] = {};
#pragma unroll
        for (int sub = 0; sub < 2; ++sub) {
            s[sub] = __builtin_amdgcn_wmma_f32_16x16x32_bf16(
                false, qa0.v, false, kf[cur].kb[sub][0].v, (short)0, s[sub],
                false, false);
            s[sub] = __builtin_amdgcn_wmma_f32_16x16x32_bf16(
                false, qa1.v, false, kf[cur].kb[sub][1].v, (short)0, s[sub],
                false, false);
        }

        // ---- scale + gated pair bias from LDS ring ----
        const float* bb = &blds[wave][cur][0];
#pragma unroll
        for (int sub = 0; sub < 2; ++sub) {
#pragma unroll
            for (int r = 0; r < 8; ++r)
                s[sub][r] = s[sub][r] * scale +
                            g * bb[(r + hf * 8) * 36 + sub * 16 + l16];
        }

        // ---- online softmax ----
#pragma unroll
        for (int r = 0; r < 8; ++r) {
            float mx = fmaxf(s[0][r], s[1][r]);
            mx = fmaxf(mx, __shfl_xor(mx, 1, 32));
            mx = fmaxf(mx, __shfl_xor(mx, 2, 32));
            mx = fmaxf(mx, __shfl_xor(mx, 4, 32));
            mx = fmaxf(mx, __shfl_xor(mx, 8, 32));
            const float mn = fmaxf(mrow[r], mx);
            const float alpha = __expf(mrow[r] - mn);
            const float p0 = __expf(s[0][r] - mn);
            const float p1 = __expf(s[1][r] - mn);
            float rs = p0 + p1;
            rs += __shfl_xor(rs, 1, 32);
            rs += __shfl_xor(rs, 2, 32);
            rs += __shfl_xor(rs, 4, 32);
            rs += __shfl_xor(rs, 8, 32);
            lrow[r] = lrow[r] * alpha + rs;
            mrow[r] = mn;
            s[0][r] = p0;
            s[1][r] = p1;
#pragma unroll
            for (int t = 0; t < 4; ++t) acc[t][r] *= alpha;
        }

        // ---- stage P via LDS into A-fragment layout ----
#pragma unroll
        for (int r = 0; r < 8; ++r) {
            const int rr = r + hf * 8;
            pw[rr * 32 + l16] = f32_to_bf16(s[0][r]);
            pw[rr * 32 + 16 + l16] = f32_to_bf16(s[1][r]);
        }
        BFrag pa;
        pa.h[0] = *(const v8u16*)(pw + l16 * 32 + hf * 16);
        pa.h[1] = *(const v8u16*)(pw + l16 * 32 + hf * 16 + 8);

        // ---- acc += P @ V_tile ----
#pragma unroll
        for (int t = 0; t < 4; ++t)
            acc[t] = __builtin_amdgcn_wmma_f32_16x16x32_bf16(
                false, pa.v, false, vb[t].v, (short)0, acc[t], false, false);
    }

    // ---- normalize, write head's 64-col slice ----
#pragma unroll
    for (int t = 0; t < 4; ++t) {
        const int d = t * 16 + l16;
#pragma unroll
        for (int r = 0; r < 8; ++r) {
            const int row = i0 + r + hf * 8;
            attn_out[(size_t)row * D_MODEL + head * HEAD_DIM + d] =
                f32_to_bf16(acc[t][r] / lrow[r]);
        }
    }
}

// ---------------------------------------------------------------------------
// Stage 4: output projection. out[2048,1024] = attn(bf16) @ Wout^T + b_out.
// ---------------------------------------------------------------------------
__global__ __launch_bounds__(128) void out_proj_kernel(
    const u16* __restrict__ ab, const u16* __restrict__ wb,
    const float* __restrict__ bias, float* __restrict__ out) {
    const int lane = threadIdx.x & 31;
    const int wave = threadIdx.x >> 5;
    const int hf = lane >> 4;
    const int l16 = lane & 15;
    const int row0 = blockIdx.y * 64 + wave * 16;
    const int col0 = blockIdx.x * 64;

    v8f acc[4] = {};
    gemm_row16_col64(ab + (size_t)(row0 + l16) * D_MODEL, wb, col0, l16, hf,
                     acc);

#pragma unroll
    for (int t = 0; t < 4; ++t) {
        const int col = col0 + t * 16 + l16;
        const float bv = bias[col];
#pragma unroll
        for (int r = 0; r < 8; ++r) {
            const int row = row0 + r + hf * 8;
            out[(size_t)row * D_MODEL + col] = acc[t][r] + bv;
        }
    }
}

// ---------------------------------------------------------------------------
extern "C" void kernel_launch(void* const* d_in, const int* in_sizes, int n_in,
                              void* d_out, int out_size, void* d_ws,
                              size_t ws_size, hipStream_t stream) {
    const float* x = (const float*)d_in[0];
    const float* pair_bias = (const float*)d_in[1];
    const float* gate = (const float*)d_in[2];
    const float* W_qkv = (const float*)d_in[3];
    const float* b_qkv = (const float*)d_in[4];
    const float* W_out = (const float*)d_in[5];
    const float* b_out = (const float*)d_in[6];
    float* out = (float*)d_out;

    char* ws = (char*)d_ws;
    size_t off = 0;
    u16* xb = (u16*)(ws + off);    off += (size_t)SEQ_L * D_MODEL * 2;
    u16* wqkvb = (u16*)(ws + off); off += (size_t)3 * D_MODEL * D_MODEL * 2;
    u16* woutb = (u16*)(ws + off); off += (size_t)D_MODEL * D_MODEL * 2;
    u16* qh = (u16*)(ws + off);    off += (size_t)N_HEADS * SEQ_L * HEAD_DIM * 2;
    u16* kh = (u16*)(ws + off);    off += (size_t)N_HEADS * SEQ_L * HEAD_DIM * 2;
    u16* vT = (u16*)(ws + off);    off += (size_t)N_HEADS * HEAD_DIM * SEQ_L * 2;
    u16* attnb = (u16*)(ws + off); off += (size_t)SEQ_L * D_MODEL * 2;

    convert_f32_to_bf16<<<512, 256, 0, stream>>>(x, xb, SEQ_L * D_MODEL);
    convert_f32_to_bf16<<<1024, 256, 0, stream>>>(W_qkv, wqkvb,
                                                  3 * D_MODEL * D_MODEL);
    convert_f32_to_bf16<<<512, 256, 0, stream>>>(W_out, woutb,
                                                 D_MODEL * D_MODEL);

    qkv_gemm_kernel<<<dim3(3 * D_MODEL / 64, SEQ_L / 64), 128, 0, stream>>>(
        xb, wqkvb, b_qkv, qh, kh, vT);

    attn_kernel<<<dim3(SEQ_L / 64, N_HEADS), 128, 0, stream>>>(
        qh, kh, vT, pair_bias, gate, attnb);

    out_proj_kernel<<<dim3(D_MODEL / 64, SEQ_L / 64), 128, 0, stream>>>(
        attnb, woutb, b_out, out);
}
